// DQNAtt_90520730730819
// MI455X (gfx1250) — matrix operand order repaired and checked by reading, not compiled, not run on hardware
//
#include <hip/hip_runtime.h>
#include <hip/hip_bf16.h>
#include <math.h>

// ---------------------------------------------------------------------------
// DQN transformer forward for MI455X (gfx1250, wave32).
// All matmuls -> v_wmma_f32_16x16x32_f16 (f16 in, f32 accum).
// Weights staged per-block into LDS via GLOBAL_LOAD_ASYNC_TO_LDS (ASYNCcnt)
// when the toolchain exposes the builtins; plain LDS staging otherwise.
// Attention   -> register flash-attention (S^T = K*Q^T trick so softmax
//                stats are per-lane scalars), V transposed through LDS.
// ---------------------------------------------------------------------------

#define HDIM   256
#define BATCH  8
#define NHEADS 8
#define DHEAD  32
#define DFFN   1024
#define TREAL  2081
#define TPAD   2112          // 132 * 16
#define SSTR   2178          // state row stride
#define NEGV   -1000000000.0f

typedef __attribute__((ext_vector_type(16))) _Float16     v16h;
typedef __attribute__((ext_vector_type(8)))  float        v8f;
typedef __attribute__((ext_vector_type(4)))  unsigned int u32x4;
typedef __attribute__((ext_vector_type(4)))  int          i32x4;

union FragH { v16h v; u32x4 q[2]; };

#if defined(__HIP_DEVICE_COMPILE__) && \
    __has_builtin(__builtin_amdgcn_global_load_async_to_lds_b128) && \
    __has_builtin(__builtin_amdgcn_s_wait_asynccnt)
#define ASYNC_LDS 1
typedef __attribute__((address_space(1))) i32x4* g_v4p;   // global int4*
typedef __attribute__((address_space(3))) i32x4* l_v4p;   // LDS int4*
#else
#define ASYNC_LDS 0
#endif

// ---------------------------------------------------------------------------
// GEMM: out[M,Nc] = act(alpha * A[M,Kpad] @ Wt[Nc,Kpad]^T + bias)
// A f16 row-major (stride Kpad); Wt f16 K-major per output column.
// Block = 128 threads = 4 waves = 4 row tiles; NT column tiles (16 each)
// shared by all waves via an LDS-staged B panel (async copy when available).
// Output rows can be remapped per-batch to scatter into the padded stream.
// ---------------------------------------------------------------------------
template <int NT>
__global__ __launch_bounds__(128) void gemm_wmma(
    const _Float16* __restrict__ A, const _Float16* __restrict__ Wt,
    const float* __restrict__ bias,
    float* __restrict__ outF32, _Float16* __restrict__ outF16,
    int M, int Kpad, int Nc, float alpha, int relu,
    int rowsPerBatch, int outBatchStride, int outRowOff)
{
    __shared__ __align__(16) _Float16 Bs[NT * 16 * 32];   // [NT*16 cols][32 k]

    const int lane = threadIdx.x & 31;
    const int wave = threadIdx.x >> 5;
    const int mt = blockIdx.x * 4 + wave;
    const bool rowOK = (mt * 16 < M);
    const int row0 = mt * 16;
    const int c0b  = blockIdx.y * (NT * 16);
    const int n = lane & 15;
    const int h = lane >> 4;

    v8f acc[NT];
#pragma unroll
    for (int j = 0; j < NT; ++j) acc[j] = (v8f){};

    const char* Abase = (const char*)A + ((size_t)(row0 + n) * Kpad) * 2 + 16 * h;

    for (int kt = 0; kt < Kpad; kt += 32) {
        // ---- stage B panel (NT*16 cols x 32 k) into LDS, 16B chunks ----
#pragma unroll
        for (int ch = threadIdx.x; ch < NT * 64; ch += 128) {
            char* gp = (char*)Wt +
                ((size_t)(c0b + (ch >> 2)) * Kpad + kt) * 2 + (ch & 3) * 16;
#if ASYNC_LDS
            __builtin_amdgcn_global_load_async_to_lds_b128(
                (g_v4p)gp, (l_v4p)(Bs + ch * 8), 0, 0);
#else
            *(u32x4*)(Bs + ch * 8) = *(const u32x4*)gp;
#endif
        }
#if ASYNC_LDS
        __builtin_amdgcn_s_wait_asynccnt(0);
#endif
        __syncthreads();

        if (rowOK) {
            FragH fa;
            const char* ap = Abase + kt * 2;
            __builtin_prefetch(ap + 64, 0, 1);           // global_prefetch_b8
            fa.q[0] = *(const u32x4*)ap;
            fa.q[1] = *(const u32x4*)(ap + 32);
#pragma unroll
            for (int j = 0; j < NT; ++j) {
                FragH fb;
                const char* bp = (const char*)Bs + (size_t)(j * 16 + n) * 64 + 16 * h;
                fb.q[0] = *(const u32x4*)bp;
                fb.q[1] = *(const u32x4*)(bp + 32);
                acc[j] = __builtin_amdgcn_wmma_f32_16x16x32_f16(
                    false, fa.v, false, fb.v, (short)0, acc[j], false, false);
            }
        }
        __syncthreads();
    }

    if (!rowOK) return;
#pragma unroll
    for (int j = 0; j < NT; ++j) {
        int col = c0b + j * 16 + n;
        float bv = bias ? bias[col] : 0.0f;
#pragma unroll
        for (int r = 0; r < 8; ++r) {
            int row = row0 + r + 8 * h;
            float v = acc[j][r] * alpha + bv;
            if (relu) v = fmaxf(v, 0.0f);
            size_t orow = (size_t)(row / rowsPerBatch) * outBatchStride
                        + outRowOff + (row % rowsPerBatch);
            if (outF32) outF32[orow * (size_t)Nc + col] = v;
            if (outF16) outF16[orow * (size_t)Nc + col] = (_Float16)v;
        }
    }
}

// ---------------------------------------------------------------------------
// Flash attention, one wave per (16-query tile, head, batch).
// S^T = K_tile(A) @ Q^T(B)  -> all 8 C components in a lane share one query,
// so online-softmax m/l are per-lane scalars (one shfl_xor(16) to merge the
// two half-wave key groups).  O^T += V^T(A, via LDS transpose) @ P(B).
// ---------------------------------------------------------------------------
__global__ __launch_bounds__(32) void flash_attn(
    const _Float16* __restrict__ Q, const _Float16* __restrict__ K,
    const _Float16* __restrict__ V, const float* __restrict__ mask,
    _Float16* __restrict__ O, int TqPad, int TkvPad)
{
    __shared__ __align__(16) _Float16 VT[32 * 40];   // 32 d rows, stride 40 (80B)
    const int lane = threadIdx.x;
    const int n = lane & 15, h = lane >> 4;
    const int qt = blockIdx.x, head = blockIdx.y, b = blockIdx.z;

    const size_t qrow = (size_t)b * TqPad + qt * 16 + n;
    FragH fq;  // Q^T B-fragment: col = query n, dh 8h..8h+7 and 16+8h..16+8h+7
    {
        const char* qp = (const char*)Q + (qrow * HDIM + head * DHEAD + h * 8) * 2;
        fq.q[0] = *(const u32x4*)qp;
        fq.q[1] = *(const u32x4*)(qp + 32);
    }

    float m_i = -3.0e38f, l_i = 0.0f;
    v8f ot0 = {}, ot1 = {};
    const float* mb = mask + (size_t)b * TkvPad;
    const size_t kvbase = (size_t)b * TkvPad;

    for (int k0 = 0; k0 < TkvPad; k0 += 32) {
        FragH fk0, fk1;  // K A-fragments: row = key, K-window = head dims
        {
            const char* kp = (const char*)K + ((kvbase + k0 + n) * HDIM + head * DHEAD) * 2 + 16 * h;
            fk0.q[0] = *(const u32x4*)kp;
            fk0.q[1] = *(const u32x4*)(kp + 32);
        }
        {
            const char* kp = (const char*)K + ((kvbase + k0 + 16 + n) * HDIM + head * DHEAD) * 2 + 16 * h;
            fk1.q[0] = *(const u32x4*)kp;
            fk1.q[1] = *(const u32x4*)(kp + 32);
        }
        v8f zero = {};
        v8f s0 = __builtin_amdgcn_wmma_f32_16x16x32_f16(false, fk0.v, false, fq.v, (short)0, zero, false, false);
        v8f s1 = __builtin_amdgcn_wmma_f32_16x16x32_f16(false, fk1.v, false, fq.v, (short)0, zero, false, false);

        float tm = -3.0e38f;
#pragma unroll
        for (int r = 0; r < 8; ++r) {
            s0[r] += mb[k0 + r + 8 * h];
            s1[r] += mb[k0 + 16 + r + 8 * h];
            tm = fmaxf(tm, fmaxf(s0[r], s1[r]));
        }
        tm = fmaxf(tm, __shfl_xor(tm, 16, 32));
        float m_new = fmaxf(m_i, tm);
        float scale = __expf(m_i - m_new);
        float tsum = 0.0f;
#pragma unroll
        for (int r = 0; r < 8; ++r) {
            s0[r] = __expf(s0[r] - m_new);
            s1[r] = __expf(s1[r] - m_new);
            tsum += s0[r] + s1[r];
        }
        tsum += __shfl_xor(tsum, 16, 32);
        l_i = l_i * scale + tsum;
        m_i = m_new;
#pragma unroll
        for (int r = 0; r < 8; ++r) { ot0[r] *= scale; ot1[r] *= scale; }

        // Re-lay P (C-layout) into B-fragment [32 keys x 16 queries]
        FragH fp;
#pragma unroll
        for (int r = 0; r < 8; ++r) {
            float o0 = __shfl_xor(s0[r], 16, 32);
            float o1 = __shfl_xor(s1[r], 16, 32);
            fp.v[r]     = (h == 0) ? (_Float16)s0[r] : (_Float16)o1;
            fp.v[8 + r] = (h == 0) ? (_Float16)o0    : (_Float16)s1[r];
        }

        // Transpose V tile (32 keys x 32 dh) through LDS -> V^T A-fragments
        {
            const u32x4* vp = (const u32x4*)((const char*)V +
                ((kvbase + k0 + lane) * HDIM + head * DHEAD) * 2);
            union { u32x4 q[4]; _Float16 e[32]; } vr;
            vr.q[0] = vp[0]; vr.q[1] = vp[1]; vr.q[2] = vp[2]; vr.q[3] = vp[3];
            __syncthreads();
#pragma unroll
            for (int d = 0; d < 32; ++d) VT[d * 40 + lane] = vr.e[d];
        }
        __syncthreads();
        FragH fv0, fv1;
        {
            const char* p = (const char*)VT + (n * 40) * 2 + 32 * h;
            fv0.q[0] = *(const u32x4*)p;
            fv0.q[1] = *(const u32x4*)(p + 16);
        }
        {
            const char* p = (const char*)VT + ((n + 16) * 40) * 2 + 32 * h;
            fv1.q[0] = *(const u32x4*)p;
            fv1.q[1] = *(const u32x4*)(p + 16);
        }
        ot0 = __builtin_amdgcn_wmma_f32_16x16x32_f16(false, fv0.v, false, fp.v, (short)0, ot0, false, false);
        ot1 = __builtin_amdgcn_wmma_f32_16x16x32_f16(false, fv1.v, false, fp.v, (short)0, ot1, false, false);
    }

    float inv = 1.0f / l_i;
    _Float16* ob = O + (qrow * HDIM + head * DHEAD);
#pragma unroll
    for (int r = 0; r < 8; ++r) {
        ob[r + 8 * h]      = (_Float16)(ot0[r] * inv);
        ob[16 + r + 8 * h] = (_Float16)(ot1[r] * inv);
    }
}

// ---------------------------------------------------------------------------
// x = LayerNorm(x + t) * g + b ; also emit f16 copy.  One block per row.
// ---------------------------------------------------------------------------
__global__ __launch_bounds__(256) void add_layernorm(
    float* __restrict__ x, const float* __restrict__ t,
    const float* __restrict__ g, const float* __restrict__ bb,
    _Float16* __restrict__ xh)
{
    __shared__ float red[18];
    int row = blockIdx.x, c = threadIdx.x;
    size_t idx = (size_t)row * HDIM + c;
    float v = x[idx] + t[idx];
    float s = v, s2 = v * v;
#pragma unroll
    for (int o = 16; o > 0; o >>= 1) { s += __shfl_xor(s, o, 32); s2 += __shfl_xor(s2, o, 32); }
    if ((c & 31) == 0) { red[c >> 5] = s; red[8 + (c >> 5)] = s2; }
    __syncthreads();
    if (c == 0) {
        float ts = 0.f, ts2 = 0.f;
        for (int i = 0; i < 8; ++i) { ts += red[i]; ts2 += red[8 + i]; }
        float mu = ts / HDIM;
        red[16] = mu; red[17] = ts2 / HDIM - mu * mu;
    }
    __syncthreads();
    float nv = (v - red[16]) * rsqrtf(red[17] + 1e-5f) * g[c] + bb[c];
    x[idx]  = nv;
    xh[idx] = (_Float16)nv;
}

// ------------------------- helpers / builders ------------------------------
__global__ void convert_weight(const float* __restrict__ W, _Float16* __restrict__ Wt,
                               int K, int N, int Kpad, int Nc)
{
    int idx = blockIdx.x * 256 + threadIdx.x;
    if (idx >= Nc * Kpad) return;
    int nn = idx / Kpad, k = idx - nn * Kpad;
    float v = (nn < N && k < K) ? W[(size_t)k * N + nn] : 0.0f;
    Wt[idx] = (_Float16)v;
}

__global__ void fill_f32(float* p, float v, size_t cnt)
{ size_t i = (size_t)blockIdx.x * 256 + threadIdx.x; if (i < cnt) p[i] = v; }

__global__ void fill_f16(_Float16* p, size_t cnt)
{ size_t i = (size_t)blockIdx.x * 256 + threadIdx.x; if (i < cnt) p[i] = (_Float16)0.0f; }

__global__ __launch_bounds__(256) void build_ns_features(
    const float* __restrict__ state, const float* __restrict__ sku,
    const float* __restrict__ nodeE, _Float16* __restrict__ X, int quantOff)
{
    int row = blockIdx.x;                 // b*1024 + node*32 + s
    int b = row >> 10, i = row & 1023;
    int node = i >> 5, s = i & 31;
    const float* sr = state + (size_t)b * SSTR;
    for (int c = threadIdx.x; c < 544; c += 256) {
        float v;
        if (c < 256)       v = sku[s * 256 + c];
        else if (c == 256) v = sr[quantOff + i];
        else if (c < 513)  v = nodeE[node * 256 + (c - 257)];
        else if (c < 515)  v = sr[1024 + node * 2 + (c - 513)];
        else               v = 0.0f;
        X[(size_t)row * 544 + c] = (_Float16)v;
    }
}

__global__ __launch_bounds__(256) void build_demand_features(
    const float* __restrict__ state, const float* __restrict__ sku,
    const float* __restrict__ demE, _Float16* __restrict__ X)
{
    int row = blockIdx.x;                 // b*32 + s
    int b = row >> 5, s = row & 31;
    const float* sr = state + (size_t)b * SSTR;
    for (int c = threadIdx.x; c < 544; c += 256) {
        float v;
        if (c < 256)       v = sku[s * 256 + c];
        else if (c == 256) v = sr[1088 + s];
        else if (c < 513)  v = demE[c - 257];          // demand_emb row 0
        else if (c < 515)  v = sr[1120 + (c - 513)];
        else               v = 0.0f;
        X[(size_t)row * 544 + c] = (_Float16)v;
    }
}

__global__ __launch_bounds__(256) void build_prod_features(
    const float* __restrict__ state, const float* __restrict__ sku,
    const float* __restrict__ demE, _Float16* __restrict__ X)
{
    int row = blockIdx.x;                 // 0..15 (rows >=8 are zero pad)
    for (int c = threadIdx.x; c < 512; c += 256) {
        float v = 0.0f;
        if (row < 8) {
            const float* sr = state + (size_t)row * SSTR;
            int cur = 0; float best = sr[2146];
            for (int s = 1; s < 32; ++s) { float x = sr[2146 + s]; if (x > best) { best = x; cur = s; } }
            v = (c < 256) ? sku[cur * 256 + c] : demE[256 + (c - 256)];   // demand_emb row 1
        }
        X[(size_t)row * 512 + c] = (_Float16)v;
    }
}

__global__ void scatter_prod(const float* __restrict__ P, float* __restrict__ x,
                             _Float16* __restrict__ xh)
{
    int b = blockIdx.x, c = threadIdx.x;
    float v = P[(size_t)b * 256 + c];
    size_t r = ((size_t)b * TPAD + 1056) * HDIM + c;
    x[r] = v; xh[r] = (_Float16)v;
}

__global__ __launch_bounds__(256) void build_enc_mask(const float* __restrict__ state,
                                                      float* __restrict__ mask)
{
    int b = blockIdx.x;
    const float* sr = state + (size_t)b * SSTR;
    for (int j = threadIdx.x; j < TPAD; j += 256) {
        float m;
        if (j < 1024)      m = (sr[j] == 0.0f) ? NEGV : 0.0f;
        else if (j < 1056) m = (sr[1088 + (j - 1024)] == 0.0f) ? NEGV : 0.0f;
        else if (j == 1056) m = 0.0f;
        else if (j < 2081) m = (sr[1122 + (j - 1057)] == 0.0f) ? NEGV : 0.0f;
        else               m = NEGV;                    // padded keys
        mask[(size_t)b * TPAD + j] = m;
    }
}

__global__ void build_y(const float* __restrict__ nodeE, float* __restrict__ y,
                        _Float16* __restrict__ yh)
{
    int row = blockIdx.x, c = threadIdx.x;              // row = b*32 + n
    float v = nodeE[(row & 31) * 256 + c];
    y[(size_t)row * 256 + c]  = v;
    yh[(size_t)row * 256 + c] = (_Float16)v;
}

__global__ void final_out(const float* __restrict__ t3, const float* __restrict__ q2b,
                          float* __restrict__ out)
{
    int i = blockIdx.x * 256 + threadIdx.x;
    if (i < 256) out[i] = t3[(size_t)i * 16] + q2b[0];
}

// ---------------------------------------------------------------------------
extern "C" void kernel_launch(void* const* d_in, const int* in_sizes, int n_in,
                              void* d_out, int out_size, void* d_ws, size_t ws_size,
                              hipStream_t stream)
{
    (void)in_sizes; (void)n_in; (void)out_size; (void)ws_size;
    const float* state = (const float*)d_in[0];
    const float* sku   = (const float*)d_in[1];
    const float* nodeE = (const float*)d_in[2];
    const float* demE  = (const float*)d_in[3];

    size_t off = 0;
    auto alloc = [&](size_t bytes) -> void* {
        void* p = (char*)d_ws + off;
        off += (bytes + 255) & ~(size_t)255;
        return p;
    };
    auto cvt = [&](int idx, int K, int N, int Kpad, int Nc) -> _Float16* {
        _Float16* wt = (_Float16*)alloc((size_t)Nc * Kpad * 2);
        int total = Nc * Kpad;
        convert_weight<<<(total + 255) / 256, 256, 0, stream>>>(
            (const float*)d_in[idx], wt, K, N, Kpad, Nc);
        return wt;
    };

    // ---- weight conversion (f32 -> transposed f16, K padded) ----
    _Float16* ns1 = cvt(4, 515, 256, 544, 256);
    _Float16* ns2 = cvt(6, 256, 256, 256, 256);
    _Float16* pe1 = cvt(8, 512, 256, 512, 256);
    _Float16* pe2 = cvt(10, 256, 256, 256, 256);
    _Float16 *eWq[2], *eWk[2], *eWv[2], *eWo[2], *eW1[2], *eW2[2];
    for (int l = 0; l < 2; ++l) {
        int ba = 12 + l * 12;
        eWq[l] = cvt(ba + 0, 256, 256, 256, 256);
        eWk[l] = cvt(ba + 1, 256, 256, 256, 256);
        eWv[l] = cvt(ba + 2, 256, 256, 256, 256);
        eWo[l] = cvt(ba + 3, 256, 256, 256, 256);
        eW1[l] = cvt(ba + 6, 256, 1024, 256, 1024);
        eW2[l] = cvt(ba + 8, 1024, 256, 1024, 256);
    }
    _Float16 *sWq[2], *sWk[2], *sWv[2], *sWo[2];
    _Float16 *cWq[2], *cWk[2], *cWv[2], *cWo[2], *dW1[2], *dW2[2];
    for (int l = 0; l < 2; ++l) {
        int ba = 36 + l * 18;
        sWq[l] = cvt(ba + 0, 256, 256, 256, 256);
        sWk[l] = cvt(ba + 1, 256, 256, 256, 256);
        sWv[l] = cvt(ba + 2, 256, 256, 256, 256);
        sWo[l] = cvt(ba + 3, 256, 256, 256, 256);
        cWq[l] = cvt(ba + 6, 256, 256, 256, 256);
        cWk[l] = cvt(ba + 7, 256, 256, 256, 256);
        cWv[l] = cvt(ba + 8, 256, 256, 256, 256);
        cWo[l] = cvt(ba + 9, 256, 256, 256, 256);
        dW1[l] = cvt(ba + 12, 256, 1024, 256, 1024);
        dW2[l] = cvt(ba + 14, 1024, 256, 1024, 256);
    }
    _Float16* q1w = cvt(72, 256, 256, 256, 256);
    _Float16* qhw = cvt(74, 256, 256, 256, 256);
    _Float16* q2w = cvt(76, 256, 1, 256, 16);

    // ---- activation workspace ----
    const size_t ME = (size_t)BATCH * TPAD;                 // 16896 rows
    _Float16* Xfeat = (_Float16*)alloc(8192ull * 544 * 2);
    _Float16* Hf    = (_Float16*)alloc(8192ull * 256 * 2);
    float*    xf    = (float*)   alloc(ME * 256 * 4);
    _Float16* xh    = (_Float16*)alloc(ME * 256 * 2);
    _Float16* qb    = (_Float16*)alloc(ME * 256 * 2);
    _Float16* kb    = (_Float16*)alloc(ME * 256 * 2);
    _Float16* vb    = (_Float16*)alloc(ME * 256 * 2);
    _Float16* ab    = (_Float16*)alloc(ME * 256 * 2);
    float*    tmp   = (float*)   alloc(ME * 256 * 4);
    _Float16* hbig  = (_Float16*)alloc(ME * 1024 * 2);
    _Float16* Hp    = (_Float16*)alloc(16ull * 256 * 2);
    float*    Pout  = (float*)   alloc(16ull * 256 * 4);
    float*    encM  = (float*)   alloc((size_t)BATCH * TPAD * 4);
    float*    decM  = (float*)   alloc((size_t)BATCH * 32 * 4);
    float*    yf    = (float*)   alloc(256ull * 256 * 4);
    _Float16* yh    = (_Float16*)alloc(256ull * 256 * 2);
    _Float16* yqb   = (_Float16*)alloc(256ull * 256 * 2);
    _Float16* ykb   = (_Float16*)alloc(256ull * 256 * 2);
    _Float16* yvb   = (_Float16*)alloc(256ull * 256 * 2);
    _Float16* yab   = (_Float16*)alloc(256ull * 256 * 2);
    _Float16* t1    = (_Float16*)alloc(256ull * 256 * 2);
    _Float16* t2    = (_Float16*)alloc(256ull * 256 * 2);
    float*    t3    = (float*)   alloc(256ull * 16 * 4);

    auto gemm = [&](const _Float16* A, const _Float16* Wt, const float* bias,
                    float* oF, _Float16* oH, int M, int Kpad, int Nc,
                    float alpha, int relu, int rpb, int obs, int oro) {
        if (Nc >= 64) {
            dim3 g((M / 16 + 3) / 4, Nc / 64);
            gemm_wmma<4><<<g, 128, 0, stream>>>(A, Wt, bias, oF, oH, M, Kpad, Nc,
                                                alpha, relu, rpb, obs, oro);
        } else {
            dim3 g((M / 16 + 3) / 4, Nc / 16);
            gemm_wmma<1><<<g, 128, 0, stream>>>(A, Wt, bias, oF, oH, M, Kpad, Nc,
                                                alpha, relu, rpb, obs, oro);
        }
    };
    auto bia = [&](int idx) { return (const float*)d_in[idx]; };

    // ---- init padded encoder stream + masks ----
    fill_f32<<<(int)((ME * 256 + 255) / 256), 256, 0, stream>>>(xf, 0.0f, ME * 256);
    fill_f16<<<(int)((ME * 256 + 255) / 256), 256, 0, stream>>>(xh, ME * 256);
    build_enc_mask<<<BATCH, 256, 0, stream>>>(state, encM);
    fill_f32<<<1, 256, 0, stream>>>(decM, 0.0f, (size_t)BATCH * 32);

    // ---- encoder input: node/sku encoders ----
    build_ns_features<<<8192, 256, 0, stream>>>(state, sku, nodeE, Xfeat, 0);     // inv
    gemm(Xfeat, ns1, bia(5), nullptr, Hf, 8192, 544, 256, 1.f, 1, 8192, 0, 0);
    gemm(Hf, ns2, bia(7), xf, xh, 8192, 256, 256, 1.f, 0, 1024, TPAD, 0);
    build_ns_features<<<8192, 256, 0, stream>>>(state, sku, nodeE, Xfeat, 1122);  // fulfill
    gemm(Xfeat, ns1, bia(5), nullptr, Hf, 8192, 544, 256, 1.f, 1, 8192, 0, 0);
    gemm(Hf, ns2, bia(7), xf, xh, 8192, 256, 256, 1.f, 0, 1024, TPAD, 1057);
    build_demand_features<<<256, 256, 0, stream>>>(state, sku, demE, Xfeat);
    gemm(Xfeat, ns1, bia(5), nullptr, Hf, 256, 544, 256, 1.f, 1, 256, 0, 0);
    gemm(Hf, ns2, bia(7), xf, xh, 256, 256, 256, 1.f, 0, 32, TPAD, 1024);
    build_prod_features<<<16, 256, 0, stream>>>(state, sku, demE, Xfeat);
    gemm(Xfeat, pe1, bia(9), nullptr, Hp, 16, 512, 256, 1.f, 1, 16, 0, 0);
    gemm(Hp, pe2, bia(11), Pout, nullptr, 16, 256, 256, 1.f, 0, 16, 0, 0);
    scatter_prod<<<BATCH, 256, 0, stream>>>(Pout, xf, xh);

    const float rs = 1.0f / sqrtf((float)DHEAD);
    const int MEi = (int)ME;

    // ---- encoder layers ----
    for (int l = 0; l < 2; ++l) {
        int ba = 12 + l * 12;
        gemm(xh, eWq[l], nullptr, nullptr, qb, MEi, 256, 256, rs,  0, MEi, 0, 0);
        gemm(xh, eWk[l], nullptr, nullptr, kb, MEi, 256, 256, 1.f, 0, MEi, 0, 0);
        gemm(xh, eWv[l], nullptr, nullptr, vb, MEi, 256, 256, 1.f, 0, MEi, 0, 0);
        flash_attn<<<dim3(TPAD / 16, NHEADS, BATCH), 32, 0, stream>>>(
            qb, kb, vb, encM, ab, TPAD, TPAD);
        gemm(ab, eWo[l], nullptr, tmp, nullptr, MEi, 256, 256, 1.f, 0, MEi, 0, 0);
        add_layernorm<<<MEi, 256, 0, stream>>>(xf, tmp, bia(ba + 4), bia(ba + 5), xh);
        gemm(xh, eW1[l], bia(ba + 7), nullptr, hbig, MEi, 256, 1024, 1.f, 1, MEi, 0, 0);
        gemm(hbig, eW2[l], bia(ba + 9), tmp, nullptr, MEi, 1024, 256, 1.f, 0, MEi, 0, 0);
        add_layernorm<<<MEi, 256, 0, stream>>>(xf, tmp, bia(ba + 10), bia(ba + 11), xh);
    }

    // ---- decoder ----
    build_y<<<256, 256, 0, stream>>>(nodeE, yf, yh);
    for (int l = 0; l < 2; ++l) {
        int ba = 36 + l * 18;
        // self attention (32 queries/batch, no mask)
        gemm(yh, sWq[l], nullptr, nullptr, yqb, 256, 256, 256, rs,  0, 256, 0, 0);
        gemm(yh, sWk[l], nullptr, nullptr, ykb, 256, 256, 256, 1.f, 0, 256, 0, 0);
        gemm(yh, sWv[l], nullptr, nullptr, yvb, 256, 256, 256, 1.f, 0, 256, 0, 0);
        flash_attn<<<dim3(2, NHEADS, BATCH), 32, 0, stream>>>(
            yqb, ykb, yvb, decM, yab, 32, 32);
        gemm(yab, sWo[l], nullptr, tmp, nullptr, 256, 256, 256, 1.f, 0, 256, 0, 0);
        add_layernorm<<<256, 256, 0, stream>>>(yf, tmp, bia(ba + 4), bia(ba + 5), yh);
        // cross attention against encoder stream
        gemm(yh, cWq[l], nullptr, nullptr, yqb, 256, 256, 256, rs,  0, 256, 0, 0);
        gemm(xh, cWk[l], nullptr, nullptr, kb, MEi, 256, 256, 1.f, 0, MEi, 0, 0);
        gemm(xh, cWv[l], nullptr, nullptr, vb, MEi, 256, 256, 1.f, 0, MEi, 0, 0);
        flash_attn<<<dim3(2, NHEADS, BATCH), 32, 0, stream>>>(
            yqb, kb, vb, encM, yab, 32, TPAD);
        gemm(yab, cWo[l], nullptr, tmp, nullptr, 256, 256, 256, 1.f, 0, 256, 0, 0);
        add_layernorm<<<256, 256, 0, stream>>>(yf, tmp, bia(ba + 10), bia(ba + 11), yh);
        // FFN
        gemm(yh, dW1[l], bia(ba + 13), nullptr, hbig, 256, 256, 1024, 1.f, 1, 256, 0, 0);
        gemm(hbig, dW2[l], bia(ba + 15), tmp, nullptr, 256, 1024, 256, 1.f, 0, 256, 0, 0);
        add_layernorm<<<256, 256, 0, stream>>>(yf, tmp, bia(ba + 16), bia(ba + 17), yh);
    }

    // ---- Q head ----
    gemm(yh, q1w, bia(73), nullptr, t1, 256, 256, 256, 1.f, 1, 256, 0, 0);
    gemm(t1, qhw, bia(75), nullptr, t2, 256, 256, 256, 1.f, 1, 256, 0, 0);
    gemm(t2, q2w, nullptr, t3, nullptr, 256, 256, 16, 1.f, 0, 256, 0, 0);
    final_out<<<1, 256, 0, stream>>>(t3, (const float*)d_in[77], (float*)d_out);
}
